// SelfAttentionBlock_68444598829537
// MI455X (gfx1250) — compile-verified, matrix operand order checked by reading, not run on hardware
//
#include <hip/hip_runtime.h>
#include <hip/hip_bf16.h>

// ---------------- types ----------------
typedef __attribute__((ext_vector_type(16))) __bf16 v16bf;
typedef __attribute__((ext_vector_type(8)))  __bf16 v8bf;
typedef __attribute__((ext_vector_type(8)))  float  v8f;

#define B_    4
#define C_    256
#define N_    4096      // 64*64 tokens per batch image
#define HEADS 8
#define HDIM  32
#define TOKS  (B_ * N_) // 16384

// Load a WMMA 16x32 bf16 A/B fragment from a row-major [rows, ld] bf16 matrix.
// Per ISA 16-bit A-layout: lane L holds row (row0 + L%16); elements 0..7 are
// K = k0 + 8*(L/16) .. +7, elements 8..15 are K = k0 + 16 + 8*(L/16) .. +7.
// B fragments (K x 16, column n per lane) load with the same pattern from a
// matrix stored with its columns as contiguous rows (i.e. transposed weights).
// Works for both global and LDS (generic) pointers.
__device__ __forceinline__ v16bf load_frag(const __bf16* base,
                                           int ld, int row0, int k0) {
  int lane = threadIdx.x & 31;
  int l16 = lane & 15, hi = lane >> 4;
  const __bf16* p = base + (size_t)(row0 + l16) * ld + (k0 + hi * 8);
  v8bf lo = *(const v8bf*)p;
  v8bf hh = *(const v8bf*)(p + 16);
  return __builtin_shufflevector(lo, hh, 0,1,2,3,4,5,6,7,8,9,10,11,12,13,14,15);
}

__device__ __forceinline__ v8f wmma_bf16(v16bf a, v16bf b, v8f c) {
  return __builtin_amdgcn_wmma_f32_16x16x32_bf16(false, a, false, b,
                                                 (short)0, c, false, false);
}

__device__ __forceinline__ v8f vzero8() {
  v8f z = {0.f,0.f,0.f,0.f,0.f,0.f,0.f,0.f};
  return z;
}

// ---------------- kernel 1: [B,C,N] -> [B,N,C] fp32 transpose ----------------
__global__ void k_transpose_cn(const float* __restrict__ x, float* __restrict__ tok) {
  __shared__ float tile[32][33];
  int b = blockIdx.z;
  int n0 = blockIdx.x * 32;
  int c0 = blockIdx.y * 32;
  const float* xb = x + (size_t)b * C_ * N_;
  float* tb = tok + (size_t)b * N_ * C_;
#pragma unroll
  for (int j = 0; j < 32; j += 8)
    tile[threadIdx.y + j][threadIdx.x] =
        xb[(size_t)(c0 + threadIdx.y + j) * N_ + (n0 + threadIdx.x)];
  __syncthreads();
#pragma unroll
  for (int j = 0; j < 32; j += 8)
    tb[(size_t)(n0 + threadIdx.y + j) * C_ + (c0 + threadIdx.x)] =
        tile[threadIdx.x][threadIdx.y + j];
}

// ---------------- kernel 2: LayerNorm over C=256, one wave per token --------
__global__ void k_layernorm(const float* __restrict__ src,
                            const float* __restrict__ g,
                            const float* __restrict__ bb,
                            __bf16* __restrict__ dst) {
  int wave = threadIdx.x >> 5, lane = threadIdx.x & 31;
  int token = blockIdx.x * 8 + wave;
  const float* row = src + (size_t)token * C_;
  float4 a  = ((const float4*)row)[lane * 2];
  float4 b2 = ((const float4*)row)[lane * 2 + 1];
  float xv[8] = {a.x, a.y, a.z, a.w, b2.x, b2.y, b2.z, b2.w};
  float s = 0.f, ss = 0.f;
#pragma unroll
  for (int i = 0; i < 8; ++i) { s += xv[i]; ss = fmaf(xv[i], xv[i], ss); }
#pragma unroll
  for (int m = 16; m >= 1; m >>= 1) {
    s  += __shfl_xor(s,  m, 32);
    ss += __shfl_xor(ss, m, 32);
  }
  float mu  = s * (1.f / 256.f);
  float var = ss * (1.f / 256.f) - mu * mu;
  float rs  = rsqrtf(var + 1e-5f);
  v8bf o;
#pragma unroll
  for (int i = 0; i < 8; ++i) {
    int c = lane * 8 + i;
    o[i] = (__bf16)((xv[i] - mu) * rs * g[c] + bb[c]);
  }
  *(v8bf*)(dst + (size_t)token * C_ + lane * 8) = o;
}

// ---------------- kernel 3: weight fp32 [K,N] -> bf16 [N,K] -----------------
__global__ void k_wtranspose(const float* __restrict__ w, __bf16* __restrict__ wt,
                             int K, int Nn) {
  int i = blockIdx.x * 256 + threadIdx.x;
  if (i < K * Nn) {
    int k = i / Nn, n = i % Nn;
    wt[(size_t)n * K + k] = (__bf16)w[i];
  }
}

// ---------------- kernel 4: QKV GEMM [16384,256]x[256,768] ------------------
// epilogue scatters into q[bh,N,d] (pre-scaled), k[bh,N,d], vT[bh,d,N]
__global__ void k_qkv_gemm(const __bf16* __restrict__ A,
                           const __bf16* __restrict__ Bt,
                           __bf16* __restrict__ q, __bf16* __restrict__ kk,
                           __bf16* __restrict__ vt) {
  const int NCG = 12; // 768 / 64
  int wid = blockIdx.x * 8 + (threadIdx.x >> 5);
  int rowBase = (wid / NCG) * 16;
  int colGrp  = (wid % NCG) * 64;
  int lane = threadIdx.x & 31, l16 = lane & 15, hi = lane >> 4;
  v8f acc[4];
#pragma unroll
  for (int t = 0; t < 4; ++t) acc[t] = vzero8();
  for (int k0 = 0; k0 < 256; k0 += 32) {
    if (k0 + 64 < 256)
      __builtin_prefetch(A + (size_t)(rowBase + l16) * 256 + k0 + 64 + hi * 8, 0, 1);
    v16bf af = load_frag(A, 256, rowBase, k0);
#pragma unroll
    for (int t = 0; t < 4; ++t) {
      v16bf bf = load_frag(Bt, 256, colGrp + 16 * t, k0);
      acc[t] = wmma_bf16(af, bf, acc[t]);
    }
  }
  const float scale = 0.17677669529663687f; // 1/sqrt(32)
#pragma unroll
  for (int t = 0; t < 4; ++t) {
#pragma unroll
    for (int r = 0; r < 8; ++r) {
      int row = rowBase + r + 8 * hi;         // global token
      int col = colGrp + 16 * t + l16;        // 0..767
      int b = row >> 12, n = row & (N_ - 1);
      int which = col >> 8, c2 = col & 255;
      int head = c2 >> 5, d = c2 & 31;
      size_t bh = (size_t)(b * HEADS + head);
      float val = acc[t][r];
      if (which == 0)      q [(bh * N_ + n) * HDIM + d] = (__bf16)(val * scale);
      else if (which == 1) kk[(bh * N_ + n) * HDIM + d] = (__bf16)val;
      else                 vt[(bh * HDIM + d) * N_ + n] = (__bf16)val;
    }
  }
}

// ---------------- kernel 5: flash attention -------------------------------
// 8 waves/block, each wave owns a 16-row Q tile; all waves share the K/V
// stream, which is staged through double-buffered LDS tiles (2KB K + 2KB V
// per buffer) loaded cooperatively: 8x less global traffic on the hot loop.
__global__ void k_flash_attn(const __bf16* __restrict__ q,
                             const __bf16* __restrict__ kk,
                             const __bf16* __restrict__ vt,
                             __bf16* __restrict__ obf) {
  __shared__ __bf16 plds[8][16 * 32];   // per-wave P staging tile (8 KB)
  __shared__ __bf16 kbuf[2][32 * 32];   // keys j..j+31, HDIM each (2x2 KB)
  __shared__ __bf16 vbuf[2][32 * 32];   // d rows x 32 keys       (2x2 KB)
  int tid = threadIdx.x;
  int wave = tid >> 5, lane = tid & 31;
  int l16 = lane & 15, hi = lane >> 4;
  int bh = blockIdx.x >> 5;                       // 32 blocks per (b,h)
  int qrow0 = (blockIdx.x & 31) * 128 + wave * 16;
  const __bf16* qp = q  + (size_t)bh * N_ * HDIM;
  const __bf16* kp = kk + (size_t)bh * N_ * HDIM;
  const __bf16* vp = vt + (size_t)bh * HDIM * N_;
  __bf16* pl = &plds[wave][0];

  // cooperative K/V tile loader: threads 0..127 -> K, 128..255 -> V.
  // chunk = 16B (8 bf16); chunk c covers tile element range [c*8, c*8+8).
  int chunk = tid & 127;
  int crow  = chunk >> 2;          // row within 32-row tile
  int cseg  = (chunk & 3) * 8;     // element offset within row
  bool isV  = tid >= 128;
  const __bf16* gsrc = isV ? (vp + (size_t)crow * N_ + cseg)   // + j0      later
                           : (kp + (size_t)crow * HDIM + cseg);// + j0*HDIM later
  size_t gstep = isV ? 32u : 32u * HDIM;  // elements advanced per 32-key step

  // prologue: stage tile 0 into buffer 0
  {
    v8bf r0 = *(const v8bf*)gsrc;
    *(v8bf*)((isV ? vbuf[0] : kbuf[0]) + chunk * 8) = r0;
  }
  __syncthreads();

  v16bf aq = load_frag(qp, HDIM, qrow0, 0);
  v8f o0 = vzero8(), o1 = vzero8();
  float mR[8], lR[8];
#pragma unroll
  for (int r = 0; r < 8; ++r) { mR[r] = -1e30f; lR[r] = 0.f; }

  const int NITER = N_ / 32;   // 128
  for (int it = 0; it < NITER; ++it) {
    int p = it & 1;
    // prefetch next tile into registers while computing on buffer p
    v8bf nreg;
    bool have = (it + 1) < NITER;
    if (have) nreg = *(const v8bf*)(gsrc + (size_t)(it + 1) * gstep);

    const __bf16* kb = kbuf[p];
    const __bf16* vb = vbuf[p];
    v16bf bk0 = load_frag(kb, 32, 0, 0);
    v16bf bk1 = load_frag(kb, 32, 16, 0);
    v8f s0 = wmma_bf16(aq, bk0, vzero8());
    v8f s1 = wmma_bf16(aq, bk1, vzero8());
#pragma unroll
    for (int r = 0; r < 8; ++r) {
      float mx = fmaxf(s0[r], s1[r]);
#pragma unroll
      for (int m = 1; m < 16; m <<= 1) mx = fmaxf(mx, __shfl_xor(mx, m, 32));
      float mn = fmaxf(mR[r], mx);
      float alpha = __expf(mR[r] - mn);
      float p0 = __expf(s0[r] - mn);
      float p1 = __expf(s1[r] - mn);
      float rs = p0 + p1;
#pragma unroll
      for (int m = 1; m < 16; m <<= 1) rs += __shfl_xor(rs, m, 32);
      lR[r] = lR[r] * alpha + rs;
      mR[r] = mn;
      o0[r] *= alpha;
      o1[r] *= alpha;
      int rowi = r + 8 * hi;                 // D-layout row
      pl[rowi * 32 + l16]      = (__bf16)p0; // key j0 + l16
      pl[rowi * 32 + 16 + l16] = (__bf16)p1; // key j0 + 16 + l16
    }
    // D-layout -> A-layout via LDS (same-wave LDS ops stay in order)
    const __bf16* pp = pl + l16 * 32 + hi * 8;
    v8bf plo = *(const v8bf*)pp;
    v8bf phi = *(const v8bf*)(pp + 16);
    v16bf pf = __builtin_shufflevector(plo, phi,
        0,1,2,3,4,5,6,7,8,9,10,11,12,13,14,15);
    v16bf vf0 = load_frag(vb, 32, 0, 0);   // d = 0..15 rows of V^T
    v16bf vf1 = load_frag(vb, 32, 16, 0);  // d = 16..31
    o0 = wmma_bf16(pf, vf0, o0);
    o1 = wmma_bf16(pf, vf1, o1);

    // stage next tile into the other buffer, then block barrier
    if (have)
      *(v8bf*)((isV ? vbuf[p ^ 1] : kbuf[p ^ 1]) + chunk * 8) = nreg;
    __syncthreads();
  }

  int b = bh >> 3, h = bh & 7;
#pragma unroll
  for (int r = 0; r < 8; ++r) {
    float inv = 1.f / lR[r];
    int tok = qrow0 + r + 8 * hi;
    size_t base = ((size_t)(b * N_ + tok)) * C_ + h * HDIM;
    obf[base + l16]      = (__bf16)(o0[r] * inv);
    obf[base + 16 + l16] = (__bf16)(o1[r] * inv);
  }
}

// ---------------- kernel 6: proj GEMM + bias + residual -> fp32 -------------
__global__ void k_proj_gemm(const __bf16* __restrict__ A,
                            const __bf16* __restrict__ Bt,
                            const float* __restrict__ bias,
                            const float* __restrict__ resid,
                            float* __restrict__ out) {
  const int NCG = 4; // 256 / 64
  int wid = blockIdx.x * 8 + (threadIdx.x >> 5);
  int rowBase = (wid / NCG) * 16;
  int colGrp  = (wid % NCG) * 64;
  int lane = threadIdx.x & 31, l16 = lane & 15, hi = lane >> 4;
  v8f acc[4];
#pragma unroll
  for (int t = 0; t < 4; ++t) acc[t] = vzero8();
  for (int k0 = 0; k0 < 256; k0 += 32) {
    if (k0 + 64 < 256)
      __builtin_prefetch(A + (size_t)(rowBase + l16) * 256 + k0 + 64 + hi * 8, 0, 1);
    v16bf af = load_frag(A, 256, rowBase, k0);
#pragma unroll
    for (int t = 0; t < 4; ++t) {
      v16bf bf = load_frag(Bt, 256, colGrp + 16 * t, k0);
      acc[t] = wmma_bf16(af, bf, acc[t]);
    }
  }
#pragma unroll
  for (int t = 0; t < 4; ++t)
#pragma unroll
    for (int r = 0; r < 8; ++r) {
      int row = rowBase + r + 8 * hi;
      int col = colGrp + 16 * t + l16;
      out[(size_t)row * C_ + col] =
          acc[t][r] + bias[col] + resid[(size_t)row * C_ + col];
    }
}

// ---------------- kernel 7: MLP fc1 GEMM + bias + exact GELU -> bf16 --------
__global__ void k_mlp1_gemm(const __bf16* __restrict__ A,
                            const __bf16* __restrict__ Bt,
                            const float* __restrict__ bias,
                            __bf16* __restrict__ out) {
  const int NCG = 8; // 512 / 64
  int wid = blockIdx.x * 8 + (threadIdx.x >> 5);
  int rowBase = (wid / NCG) * 16;
  int colGrp  = (wid % NCG) * 64;
  int lane = threadIdx.x & 31, l16 = lane & 15, hi = lane >> 4;
  v8f acc[4];
#pragma unroll
  for (int t = 0; t < 4; ++t) acc[t] = vzero8();
  for (int k0 = 0; k0 < 256; k0 += 32) {
    if (k0 + 64 < 256)
      __builtin_prefetch(A + (size_t)(rowBase + l16) * 256 + k0 + 64 + hi * 8, 0, 1);
    v16bf af = load_frag(A, 256, rowBase, k0);
#pragma unroll
    for (int t = 0; t < 4; ++t) {
      v16bf bf = load_frag(Bt, 256, colGrp + 16 * t, k0);
      acc[t] = wmma_bf16(af, bf, acc[t]);
    }
  }
#pragma unroll
  for (int t = 0; t < 4; ++t)
#pragma unroll
    for (int r = 0; r < 8; ++r) {
      int row = rowBase + r + 8 * hi;
      int col = colGrp + 16 * t + l16;
      float u = acc[t][r] + bias[col];
      float gl = 0.5f * u * (1.f + erff(u * 0.70710678118654752f));
      out[(size_t)row * 512 + col] = (__bf16)gl;
    }
}

// ---------------- kernel 8: MLP fc2 GEMM + bias + residual, store [B,C,N] ---
__global__ void k_mlp2_gemm(const __bf16* __restrict__ A,
                            const __bf16* __restrict__ Bt,
                            const float* __restrict__ bias,
                            const float* __restrict__ resid,
                            float* __restrict__ out) {
  const int NCG = 4; // 256 / 64
  int wid = blockIdx.x * 8 + (threadIdx.x >> 5);
  int rowBase = (wid / NCG) * 16;
  int colGrp  = (wid % NCG) * 64;
  int lane = threadIdx.x & 31, l16 = lane & 15, hi = lane >> 4;
  v8f acc[4];
#pragma unroll
  for (int t = 0; t < 4; ++t) acc[t] = vzero8();
  for (int k0 = 0; k0 < 512; k0 += 32) {
    if (k0 + 64 < 512)
      __builtin_prefetch(A + (size_t)(rowBase + l16) * 512 + k0 + 64 + hi * 8, 0, 1);
    v16bf af = load_frag(A, 512, rowBase, k0);
#pragma unroll
    for (int t = 0; t < 4; ++t) {
      v16bf bf = load_frag(Bt, 512, colGrp + 16 * t, k0);
      acc[t] = wmma_bf16(af, bf, acc[t]);
    }
  }
#pragma unroll
  for (int t = 0; t < 4; ++t)
#pragma unroll
    for (int r = 0; r < 8; ++r) {
      int row = rowBase + r + 8 * hi;        // global token
      int col = colGrp + 16 * t + l16;       // channel
      float val = acc[t][r] + bias[col] + resid[(size_t)row * C_ + col];
      int b = row >> 12, n = row & (N_ - 1);
      out[((size_t)b * C_ + col) * N_ + n] = val;   // [B,C,H,W]
    }
}

// ---------------- host launcher ----------------
extern "C" void kernel_launch(void* const* d_in, const int* in_sizes, int n_in,
                              void* d_out, int out_size, void* d_ws, size_t ws_size,
                              hipStream_t stream) {
  (void)in_sizes; (void)n_in; (void)out_size; (void)ws_size;
  const float* x    = (const float*)d_in[0];
  const float* g1   = (const float*)d_in[1];
  const float* b1   = (const float*)d_in[2];
  const float* Wqkv = (const float*)d_in[3];
  const float* Wp   = (const float*)d_in[4];
  const float* bp   = (const float*)d_in[5];
  const float* g2   = (const float*)d_in[6];
  const float* b2   = (const float*)d_in[7];
  const float* W1   = (const float*)d_in[8];
  const float* bf1  = (const float*)d_in[9];
  const float* W2   = (const float*)d_in[10];
  const float* bf2  = (const float*)d_in[11];

  char* ws = (char*)d_ws;
  size_t off = 0;
  auto alloc = [&](size_t bytes) {
    void* p = ws + off;
    off += (bytes + 255) & ~(size_t)255;
    return p;
  };
  float*  tok0  = (float*) alloc((size_t)TOKS * C_ * 4);   // residual stream
  float*  tok2  = (float*) alloc((size_t)TOKS * C_ * 4);   // post-attn residual
  __bf16* t1bf  = (__bf16*)alloc((size_t)TOKS * C_ * 2);
  __bf16* t2bf  = (__bf16*)alloc((size_t)TOKS * C_ * 2);
  __bf16* qbf   = (__bf16*)alloc((size_t)B_ * HEADS * N_ * HDIM * 2);
  __bf16* kbf   = (__bf16*)alloc((size_t)B_ * HEADS * N_ * HDIM * 2);
  __bf16* vtbf  = (__bf16*)alloc((size_t)B_ * HEADS * HDIM * N_ * 2);
  __bf16* obf   = (__bf16*)alloc((size_t)TOKS * C_ * 2);
  __bf16* hbf   = (__bf16*)alloc((size_t)TOKS * 512 * 2);
  __bf16* WqkvT = (__bf16*)alloc((size_t)768 * 256 * 2);
  __bf16* WpT   = (__bf16*)alloc((size_t)256 * 256 * 2);
  __bf16* W1T   = (__bf16*)alloc((size_t)512 * 256 * 2);
  __bf16* W2T   = (__bf16*)alloc((size_t)256 * 512 * 2);

  // 1) [B,C,N] -> [B,N,C]
  k_transpose_cn<<<dim3(N_ / 32, C_ / 32, B_), dim3(32, 8), 0, stream>>>(x, tok0);
  // 2) LN1 -> bf16
  k_layernorm<<<TOKS / 8, 256, 0, stream>>>(tok0, g1, b1, t1bf);
  // 3) weight transposes (fp32 -> bf16 [N,K])
  k_wtranspose<<<(256 * 768 + 255) / 256, 256, 0, stream>>>(Wqkv, WqkvT, 256, 768);
  k_wtranspose<<<(256 * 256 + 255) / 256, 256, 0, stream>>>(Wp,   WpT,   256, 256);
  k_wtranspose<<<(256 * 512 + 255) / 256, 256, 0, stream>>>(W1,   W1T,   256, 512);
  k_wtranspose<<<(512 * 256 + 255) / 256, 256, 0, stream>>>(W2,   W2T,   512, 256);
  // 4) QKV projection (scatters into q / k / v^T, scale folded into q)
  k_qkv_gemm<<<(TOKS / 16) * 12 / 8, 256, 0, stream>>>(t1bf, WqkvT, qbf, kbf, vtbf);
  // 5) flash attention (LDS-staged, double-buffered K/V)
  k_flash_attn<<<(B_ * HEADS) * (N_ / 128), 256, 0, stream>>>(qbf, kbf, vtbf, obf);
  // 6) output projection + residual
  k_proj_gemm<<<(TOKS / 16) * 4 / 8, 256, 0, stream>>>(obf, WpT, bp, tok0, tok2);
  // 7) LN2 -> bf16
  k_layernorm<<<TOKS / 8, 256, 0, stream>>>(tok2, g2, b2, t2bf);
  // 8) MLP fc1 + GELU
  k_mlp1_gemm<<<(TOKS / 16) * 8 / 8, 256, 0, stream>>>(t2bf, W1T, bf1, hbf);
  // 9) MLP fc2 + residual, transposed store to [B,C,H,W]
  k_mlp2_gemm<<<(TOKS / 16) * 4 / 8, 256, 0, stream>>>(hbf, W2T, bf2, tok2,
                                                       (float*)d_out);
}
